// GrCNNEncoder_68616397521054
// MI455X (gfx1250) — compile-verified, matrix operand order checked by reading, not run on hardware
//
#include <hip/hip_runtime.h>

// ---------------------------------------------------------------------------
// GrCNN encoder for MI455X (gfx1250, wave32, WMMA).
//   * prep kernel: fp32 weights -> bf16, pre-swizzled into per-lane WMMA
//     B-fragment layout (frag = [ct*8+ks], 32 lanes x 16 bf16 contiguous).
//   * main kernel: 1 workgroup (8 waves) per batch. Level tensor double-
//     buffered in LDS as bf16; all GEMMs via v_wmma_f32_16x16x32_bf16 with
//     f32 accumulation; gates/blend/max-pool on VALU from LDS.
//   * all fp32<->bf16 conversion through native __bf16 casts so the backend
//     can use the gfx1250 packed-convert hardware instead of ALU emulation.
// ---------------------------------------------------------------------------

typedef __attribute__((ext_vector_type(16))) __bf16        v16bf;
typedef __attribute__((ext_vector_type(8)))  float         v8f;
typedef __attribute__((ext_vector_type(4)))  unsigned int  u32x4;
typedef __attribute__((ext_vector_type(8)))  unsigned int  u32x8;

#define SEQ  128
#define HID  256
#define NB   32

__device__ __forceinline__ unsigned short f2bf(float f) {
  __bf16 h = (__bf16)f;                       // native hw convert
  return __builtin_bit_cast(unsigned short, h);
}
__device__ __forceinline__ float bf2f(unsigned short u) {
  __bf16 h = __builtin_bit_cast(__bf16, u);
  return (float)h;                            // native hw convert
}
// pack two fp32 -> one dword of two bf16 (backend can select v_cvt_pk_bf16_f32)
__device__ __forceinline__ unsigned int pkbf2(float a, float b) {
  return (unsigned int)f2bf(a) | ((unsigned int)f2bf(b) << 16);
}
__device__ __forceinline__ v16bf frag_cast(u32x4 lo, u32x4 hi) {
  u32x8 w = __builtin_shufflevector(lo, hi, 0, 1, 2, 3, 4, 5, 6, 7);
  return __builtin_bit_cast(v16bf, w);
}
__device__ __forceinline__ v8f wmma_bf16(v16bf a, v16bf b, v8f c) {
  return __builtin_amdgcn_wmma_f32_16x16x32_bf16(false, a, false, b,
                                                 (short)0, c, false, false);
}

// A-fragment (16x32 bf16) from LDS, row-major rows at stride 256 bf16.
// Lane half g=lane>>4: chunk1 K = ks*32+g*8..+7, chunk2 K = ks*32+16+g*8..+7.
__device__ __forceinline__ v16bf load_a_lds(const unsigned short* lev, int row,
                                            int ks, int lane) {
  int g = lane >> 4;
  const unsigned short* p = lev + row * HID + ks * 32 + g * 8;
  u32x4 lo = *(const u32x4*)(p);
  u32x4 hi = *(const u32x4*)(p + 16);
  return frag_cast(lo, hi);
}

// B-fragment (32x16 bf16) from pre-swizzled global ws: 32B contiguous per lane.
__device__ __forceinline__ v16bf load_b_ws(const unsigned short* wsb, int ct,
                                           int ks, int lane) {
  const unsigned short* p = wsb + (((ct * 8 + ks) * 32 + lane) << 4);
  u32x4 lo = *(const u32x4*)(p);
  u32x4 hi = *(const u32x4*)(p + 8);
  return frag_cast(lo, hi);
}

// 8 consecutive fp32 -> 8 packed bf16 (4 packed converts)
__device__ __forceinline__ u32x4 cvt8(const float* p) {
  u32x4 r;
#pragma unroll
  for (int i = 0; i < 4; ++i) r[i] = pkbf2(p[2 * i], p[2 * i + 1]);
  return r;
}
__device__ __forceinline__ v16bf load_a_f32(const float* rowp, int ks, int lane) {
  int g = lane >> 4;
  const float* p = rowp + ks * 32 + g * 8;
  return frag_cast(cvt8(p), cvt8(p + 16));
}

// ---------------------------------------------------------------------------
// Prep: swizzle U_val, W_l, W_r, cls_W (transposed) into bf16 WMMA fragments.
// ws element index == global thread id; 4 matrices x 65536 elems.
// ---------------------------------------------------------------------------
__global__ __launch_bounds__(256) void grcnn_prep(
    const float* __restrict__ U, const float* __restrict__ Wl,
    const float* __restrict__ Wr, const float* __restrict__ cls,
    unsigned short* __restrict__ ws) {
  int t = blockIdx.x * 256 + threadIdx.x;   // 0..262143
  int m = t >> 16;                          // matrix id
  int r = t & 65535;
  int frag = r >> 9;                        // ct*8 + ks
  int lane = (r >> 4) & 31;
  int e    = r & 15;
  int ct = frag >> 3, ks = frag & 7;
  int n = ct * 16 + (lane & 15);
  int g = lane >> 4;
  int v = e >> 1, eo = e & 1;
  int k = ks * 32 + ((v < 4) ? 0 : 16) + g * 8 + (v & 3) * 2 + eo;
  float val;
  if (m == 0)      val = U[k * HID + n];
  else if (m == 1) val = Wl[k * HID + n];
  else if (m == 2) val = Wr[k * HID + n];
  else             val = cls[n * HID + k];   // transpose: B[k][n] = cls_W[n][k]
  ws[t] = f2bf(val);
}

// ---------------------------------------------------------------------------
// Main: one workgroup per batch.
// LDS: lev0[130*256], lev1[130*256], pooled[128*256] (bf16),
//      gates[128*4], G_l[768], G_r[768], G_b[4], W_b[256], cls_b[256] (fp32)
// ---------------------------------------------------------------------------
__global__ __launch_bounds__(256) void grcnn_main(
    const int* __restrict__ tokens, const float* __restrict__ emb,
    const float* __restrict__ Wb, const float* __restrict__ Gl,
    const float* __restrict__ Gr, const float* __restrict__ Gb,
    const float* __restrict__ clsb, const unsigned short* __restrict__ ws,
    float* __restrict__ out) {
  extern __shared__ char smem[];
  unsigned short* lev0   = (unsigned short*)smem;       // 130*256
  unsigned short* lev1   = lev0 + 130 * HID;            // 130*256
  unsigned short* pooled = lev1 + 130 * HID;            // 128*256
  float* gates = (float*)(pooled + SEQ * HID);          // 128*4
  float* sGl   = gates + SEQ * 4;                       // 768
  float* sGr   = sGl + 768;                             // 768
  float* sGb   = sGr + 768;                             // 4
  float* sWb   = sGb + 4;                               // 256
  float* sCb   = sWb + HID;                             // 256

  const int tid  = threadIdx.x;
  const int lane = tid & 31;
  const int wave = tid >> 5;
  const int b    = blockIdx.x;

  const unsigned short* wU  = ws;
  const unsigned short* wWl = ws + 65536;
  const unsigned short* wWr = ws + 131072;
  const unsigned short* wC  = ws + 196608;

  // stage small constants + zero the pad rows (row 128/129) of both buffers
  for (int i = tid; i < 768; i += 256) { sGl[i] = Gl[i]; sGr[i] = Gr[i]; }
  sWb[tid] = Wb[tid];
  sCb[tid] = clsb[tid];
  if (tid < 4) sGb[tid] = (tid < 3) ? Gb[tid] : 0.0f;
  for (int i = tid; i < 2 * HID; i += 256) {
    lev0[SEQ * HID + i] = 0;
    lev1[SEQ * HID + i] = 0;
  }
  // warm L2 for the scan's weight fragments (global_prefetch_b8)
  __builtin_prefetch(wWl + (size_t)tid * 512, 0, 1);
  __builtin_prefetch(wWr + (size_t)tid * 512, 0, 1);
  __syncthreads();

  // ---- Phase A: level0 = gather(embedding) @ U_val -------------------------
  {
    const int tokbase = b * SEQ;
    for (int cti = 0; cti < 2; ++cti) {
      int ct = wave * 2 + cti;
      v16bf bu[8];
#pragma unroll
      for (int ks = 0; ks < 8; ++ks) bu[ks] = load_b_ws(wU, ct, ks, lane);
      for (int rt = 0; rt < 8; ++rt) {
        int row = rt * 16 + (lane & 15);
        int tok = tokens[tokbase + row];
        const float* rowp = emb + (size_t)tok * HID;
        v8f acc = {};
#pragma unroll
        for (int ks = 0; ks < 8; ++ks) {
          v16bf a = load_a_f32(rowp, ks, lane);
          acc = wmma_bf16(a, bu[ks], acc);
        }
        int n  = ct * 16 + (lane & 15);
        int mg = (lane >> 4) * 8;
#pragma unroll
        for (int r = 0; r < 8; ++r)
          lev0[(rt * 16 + mg + r) * HID + n] = f2bf(acc[r]);
      }
    }
  }
  __syncthreads();

  // pooled[0] = max over all 128 positions
  {
    float m = -3.0e38f;
    for (int p = 0; p < SEQ; ++p) m = fmaxf(m, bf2f(lev0[p * HID + tid]));
    pooled[tid] = f2bf(m);
  }
  __syncthreads();

  // ---- Scan: 127 sequential levels ----------------------------------------
  unsigned short* cur = lev0;
  unsigned short* nxt = lev1;
  for (int j = 0; j < SEQ - 1; ++j) {
    const int L   = SEQ - j;            // valid length of current level
    const int Lm1 = L - 1;              // positions produced this step
    const int nT  = (Lm1 + 15) >> 4;    // row tiles needed (uniform)

    // gates: softmax(left@G_l + right@G_r + G_b), one thread per position
    if (tid < Lm1) {
      const unsigned short* lp = cur + tid * HID;
      const unsigned short* rp = lp + HID;
      float z0 = sGb[0], z1 = sGb[1], z2 = sGb[2];
      for (int k = 0; k < HID; k += 8) {
        u32x4 lu = *(const u32x4*)(lp + k);
        u32x4 ru = *(const u32x4*)(rp + k);
#pragma unroll
        for (int q = 0; q < 4; ++q) {
          unsigned int lw = lu[q], rw = ru[q];
          float l0 = bf2f((unsigned short)lw), l1 = bf2f((unsigned short)(lw >> 16));
          float r0 = bf2f((unsigned short)rw), r1 = bf2f((unsigned short)(rw >> 16));
          const float* gl = sGl + (k + q * 2) * 3;
          const float* gr = sGr + (k + q * 2) * 3;
          z0 += l0 * gl[0] + l1 * gl[3] + r0 * gr[0] + r1 * gr[3];
          z1 += l0 * gl[1] + l1 * gl[4] + r0 * gr[1] + r1 * gr[4];
          z2 += l0 * gl[2] + l1 * gl[5] + r0 * gr[2] + r1 * gr[5];
        }
      }
      float mz = fmaxf(z0, fmaxf(z1, z2));
      float e0 = __expf(z0 - mz), e1 = __expf(z1 - mz), e2 = __expf(z2 - mz);
      float inv = 1.0f / (e0 + e1 + e2);
      gates[tid * 4 + 0] = e0 * inv;
      gates[tid * 4 + 1] = e1 * inv;
      gates[tid * 4 + 2] = e2 * inv;
    }
    __syncthreads();

    // fused GEMM: central = relu(left@W_l + right@W_r + W_b); blend; store nxt
    for (int cti = 0; cti < 2; ++cti) {
      int ct = wave * 2 + cti;
      v16bf bl[8], br[8];
#pragma unroll
      for (int ks = 0; ks < 8; ++ks) {
        bl[ks] = load_b_ws(wWl, ct, ks, lane);
        br[ks] = load_b_ws(wWr, ct, ks, lane);
      }
      for (int rt = 0; rt < nT; ++rt) {
        int base = rt * 16 + (lane & 15);
        v8f acc = {};
#pragma unroll
        for (int ks = 0; ks < 8; ++ks) {
          v16bf al = load_a_lds(cur, base,     ks, lane);
          v16bf ar = load_a_lds(cur, base + 1, ks, lane);
          acc = wmma_bf16(al, bl[ks], acc);
          acc = wmma_bf16(ar, br[ks], acc);
        }
        int n  = ct * 16 + (lane & 15);
        int mg = (lane >> 4) * 8;
        float wb = sWb[n];
#pragma unroll
        for (int r = 0; r < 8; ++r) {
          int p = rt * 16 + mg + r;
          float central = fmaxf(acc[r] + wb, 0.0f);
          float g0 = gates[p * 4 + 0];
          float g1 = gates[p * 4 + 1];
          float g2 = gates[p * 4 + 2];
          float lf = bf2f(cur[p * HID + n]);
          float rf = bf2f(cur[(p + 1) * HID + n]);
          nxt[p * HID + n] = f2bf(g0 * lf + g1 * central + g2 * rf);
        }
      }
    }
    __syncthreads();

    // pooled[j+1] = max over valid positions of the new level
    {
      float m = -3.0e38f;
      for (int p = 0; p < Lm1; ++p) m = fmaxf(m, bf2f(nxt[p * HID + tid]));
      pooled[(j + 1) * HID + tid] = f2bf(m);
    }
    unsigned short* t = cur; cur = nxt; nxt = t;
    __syncthreads();
  }

  // ---- Classifier: relu(pooled @ cls_W^T + cls_b) -> out[b, level, cls] ----
  for (int cti = 0; cti < 2; ++cti) {
    int ct = wave * 2 + cti;
    v16bf bc[8];
#pragma unroll
    for (int ks = 0; ks < 8; ++ks) bc[ks] = load_b_ws(wC, ct, ks, lane);
    for (int rt = 0; rt < 8; ++rt) {
      int base = rt * 16 + (lane & 15);
      v8f acc = {};
#pragma unroll
      for (int ks = 0; ks < 8; ++ks) {
        v16bf a = load_a_lds(pooled, base, ks, lane);
        acc = wmma_bf16(a, bc[ks], acc);
      }
      int n  = ct * 16 + (lane & 15);
      int mg = (lane >> 4) * 8;
      float cb = sCb[n];
      float* outp = out + (size_t)b * SEQ * HID;
#pragma unroll
      for (int r = 0; r < 8; ++r) {
        int p = rt * 16 + mg + r;
        outp[(size_t)p * HID + n] = fmaxf(acc[r] + cb, 0.0f);
      }
    }
  }
}

// ---------------------------------------------------------------------------
extern "C" void kernel_launch(void* const* d_in, const int* in_sizes, int n_in,
                              void* d_out, int out_size, void* d_ws,
                              size_t ws_size, hipStream_t stream) {
  (void)in_sizes; (void)n_in; (void)out_size; (void)ws_size;
  const int*   tokens = (const int*)d_in[0];
  const float* emb    = (const float*)d_in[1];
  const float* U      = (const float*)d_in[2];
  const float* Wl     = (const float*)d_in[3];
  const float* Wr     = (const float*)d_in[4];
  const float* Wb     = (const float*)d_in[5];
  const float* Gl     = (const float*)d_in[6];
  const float* Gr     = (const float*)d_in[7];
  const float* Gb     = (const float*)d_in[8];
  const float* clsW   = (const float*)d_in[9];
  const float* clsb   = (const float*)d_in[10];
  unsigned short* ws  = (unsigned short*)d_ws;
  float* out          = (float*)d_out;

  // 4 matrices x 65536 bf16 fragments = 512 KB of workspace
  grcnn_prep<<<1024, 256, 0, stream>>>(U, Wl, Wr, clsW, ws);

  const size_t smem =
      (size_t)(2 * 130 * HID + SEQ * HID) * sizeof(unsigned short) +  // levels+pooled
      (size_t)(SEQ * 4 + 768 + 768 + 4 + HID + HID) * sizeof(float);  // gates+consts
  grcnn_main<<<NB, 256, smem, stream>>>(tokens, emb, Wb, Gl, Gr, Gb, clsb, ws,
                                        out);
}
